// HAN_60026462929256
// MI455X (gfx1250) — matrix operand order repaired and checked by reading, not compiled, not run on hardware
//
#include <hip/hip_runtime.h>
#include <hip/hip_bf16.h>
#include <math.h>

#define NN 50000
#define NE 800000
#define IN_CH 128
#define HID 64
#define HEADS 8
#define OUT_CH 16
#define NEG_SLOPE 0.2f

typedef __attribute__((ext_vector_type(16))) __bf16 bf16x16;
typedef __attribute__((ext_vector_type(8)))  float  floatx8;

// ---------------------------------------------------------------------------
// WMMA fragment loaders (this file's contiguous k order for A/B, wave32)
// A: 16x32 bf16.  lane l: M = l&15 ; khalf = (l>=16)?16:0 ;
//    VGPR v holds K pair: khalf + 2v
// B: 32x16 bf16 (B[k][n] = W[n][k]).  lane l: N = l&15 ; khalf = (l>=16)?16:0 ;
//    VGPR v holds K pair: khalf + 2v
// C/D f32 16x16: lane l, vgpr r -> M = (l>>4)*8 + r, N = l&15
// ---------------------------------------------------------------------------
__device__ inline bf16x16 frag_a(const __bf16* __restrict__ As, int ldk, int kb, int lane) {
  const int m  = lane & 15;
  const int kh = (lane >> 4) << 3;
  const __bf16* r = As + m * ldk + kb;
  bf16x16 a;
#pragma unroll
  for (int v = 0; v < 8; ++v) {
    int k = (kh << 1) + (v << 1);
    a[2 * v]     = r[k];
    a[2 * v + 1] = r[k + 1];
  }
  return a;
}

__device__ inline bf16x16 frag_b(const __bf16* __restrict__ Ws, int ldk, int kb, int ncol0, int lane) {
  const int n  = ncol0 + (lane & 15);
  const int kh = (lane >> 4) << 4;
  const __bf16* r = Ws + n * ldk + kb + kh;
  bf16x16 b;
#pragma unroll
  for (int v = 0; v < 8; ++v) {
    b[2 * v]     = r[2 * v];
    b[2 * v + 1] = r[2 * v + 1];
  }
  return b;
}

__device__ inline void atomAddF(float* p, float v) {
  __hip_atomic_fetch_add(p, v, __ATOMIC_RELAXED, __HIP_MEMORY_SCOPE_AGENT);
}

__device__ inline void atomMaxF(float* addr, float val) {
  int* ai = (int*)addr;
  int old = *ai;
  while (__int_as_float(old) < val) {
    int assumed = old;
    old = atomicCAS(ai, assumed, __float_as_int(val));
    if (old == assumed) break;
  }
}

// ---------------------------------------------------------------------------
// Projection GEMM: XP = X @ W^T + b   (N x KDIM) x (64 x KDIM) -> (N x 64)
// Fused epilogue: per-head attention scores for both edge types.
// Block = 128 threads (4 waves), tile = 16 rows x 64 cols.
// ---------------------------------------------------------------------------
template <int KDIM>
__global__ void __launch_bounds__(128)
proj_att_kernel(const float* __restrict__ X, const float* __restrict__ W,
                const float* __restrict__ bias,
                const float* __restrict__ as0, const float* __restrict__ as1,
                const float* __restrict__ ad0, const float* __restrict__ ad1,
                float* __restrict__ XP,
                float* __restrict__ Asrc0, float* __restrict__ Asrc1,
                float* __restrict__ Adst0, float* __restrict__ Adst1) {
  __shared__ __bf16 As[16 * KDIM];
  __shared__ __bf16 Ws[64 * KDIM];
  __shared__ float  Yt[16 * 64];

  const int t = threadIdx.x;
  const int lane = t & 31;
  const int wave = t >> 5;
  const int row0 = blockIdx.x * 16;

  for (int i = t; i < 16 * KDIM; i += 128) {
    int r = i / KDIM, k = i % KDIM;
    As[i] = (__bf16)X[(row0 + r) * KDIM + k];
  }
  for (int i = t; i < 64 * KDIM; i += 128) Ws[i] = (__bf16)W[i];
  __syncthreads();

  floatx8 acc = {};
#pragma unroll
  for (int kb = 0; kb < KDIM; kb += 32) {
    bf16x16 a = frag_a(As, KDIM, kb, lane);
    bf16x16 b = frag_b(Ws, KDIM, kb, wave * 16, lane);
    acc = __builtin_amdgcn_wmma_f32_16x16x32_bf16(false, a, false, b, (short)0, acc,
                                                  false, false);
  }

  const int n  = lane & 15;
  const int mo = (lane >> 4) << 3;
#pragma unroll
  for (int r = 0; r < 8; ++r)
    Yt[(mo + r) * 64 + wave * 16 + n] = acc[r] + bias[wave * 16 + n];
  __syncthreads();

#pragma unroll
  for (int i = 0; i < 8; ++i)
    XP[row0 * 64 + t * 8 + i] = Yt[t * 8 + i];

  // attention epilogue: thread -> (row = t>>3, head = t&7)
  const int rr = t >> 3, h = t & 7;
  const float* y = Yt + rr * 64 + h * 8;
  float s0 = 0.f, s1 = 0.f, d0 = 0.f, d1 = 0.f;
#pragma unroll
  for (int dd = 0; dd < 8; ++dd) {
    float v = y[dd];
    s0 += v * as0[h * 8 + dd];
    s1 += v * as1[h * 8 + dd];
    d0 += v * ad0[h * 8 + dd];
    d1 += v * ad1[h * 8 + dd];
  }
  const int o = (row0 + rr) * 8 + h;
  Asrc0[o] = s0; Asrc1[o] = s1; Adst0[o] = d0; Adst1[o] = d1;
}

// ---------------------------------------------------------------------------
// Edge kernels (segment softmax, atomics-based)
// ---------------------------------------------------------------------------
__global__ void fill_kernel(float* __restrict__ p, float v, int n) {
  int i = blockIdx.x * blockDim.x + threadIdx.x;
  if (i < n) p[i] = v;
}

__global__ void edge_max_kernel(const int* __restrict__ src, const int* __restrict__ dst,
                                const float* __restrict__ Asrc, const float* __restrict__ Adst,
                                float* __restrict__ segmax, int E) {
  int i = blockIdx.x * blockDim.x + threadIdx.x;
  if (i >= E * 8) return;
  const int e = i >> 3, h = i & 7;
  const int s = src[e], d = dst[e];
  float al = Asrc[s * 8 + h] + Adst[d * 8 + h];
  al = al > 0.f ? al : NEG_SLOPE * al;
  atomMaxF(&segmax[d * 8 + h], al);
}

// fused: e = exp(alpha - max); den += e; agg[d, h*8+dd] += e * xp[s, h*8+dd]
__global__ void edge_agg_kernel(const int* __restrict__ src, const int* __restrict__ dst,
                                const float* __restrict__ Asrc, const float* __restrict__ Adst,
                                const float* __restrict__ segmax, const float* __restrict__ XP,
                                float* __restrict__ den, float* __restrict__ agg, int E) {
  int i = blockIdx.x * blockDim.x + threadIdx.x;
  if (i >= E * 8) return;
  const int e = i >> 3, h = i & 7;
  const int s = src[e], d = dst[e];
  const float* xs = XP + s * 64 + h * 8;
  __builtin_prefetch(xs, 0, 0);  // global_prefetch_b8 (L2-resident gather)
  float al = Asrc[s * 8 + h] + Adst[d * 8 + h];
  al = al > 0.f ? al : NEG_SLOPE * al;
  const float ex = __expf(al - segmax[d * 8 + h]);
  atomAddF(&den[d * 8 + h], ex);
  float* ag = agg + d * 64 + h * 8;
#pragma unroll
  for (int dd = 0; dd < 8; ++dd) atomAddF(&ag[dd], ex * xs[dd]);
}

// out = relu(agg / den) in-place (den==0 -> 0)
__global__ void norm_relu_kernel(const float* __restrict__ den, float* __restrict__ st, int n) {
  int i = blockIdx.x * blockDim.x + threadIdx.x;
  if (i >= n) return;
  const int node = i >> 6, h = (i & 63) >> 3;
  const float dn = den[node * 8 + h];
  float v = dn > 0.f ? st[i] / dn : 0.f;
  st[i] = v > 0.f ? v : 0.f;
}

// ---------------------------------------------------------------------------
// Semantic attention: colsum[c] += sum_rows tanh(st @ kW^T + kb)[.,c]
// ---------------------------------------------------------------------------
__global__ void __launch_bounds__(128)
tanh_colsum_kernel(const float* __restrict__ ST, const float* __restrict__ KW,
                   const float* __restrict__ kbias, float* __restrict__ colsum) {
  __shared__ __bf16 As[16 * 64];
  __shared__ __bf16 Ws[64 * 64];
  __shared__ float  Yt[16 * 64];

  const int t = threadIdx.x, lane = t & 31, wave = t >> 5;
  const int row0 = blockIdx.x * 16;

  for (int i = t; i < 16 * 64; i += 128) As[i] = (__bf16)ST[row0 * 64 + i];
  for (int i = t; i < 64 * 64; i += 128) Ws[i] = (__bf16)KW[i];
  __syncthreads();

  floatx8 acc = {};
#pragma unroll
  for (int kb = 0; kb < 64; kb += 32) {
    bf16x16 a = frag_a(As, 64, kb, lane);
    bf16x16 b = frag_b(Ws, 64, kb, wave * 16, lane);
    acc = __builtin_amdgcn_wmma_f32_16x16x32_bf16(false, a, false, b, (short)0, acc,
                                                  false, false);
  }
  const int n = lane & 15, mo = (lane >> 4) << 3;
#pragma unroll
  for (int r = 0; r < 8; ++r)
    Yt[(mo + r) * 64 + wave * 16 + n] = tanhf(acc[r] + kbias[wave * 16 + n]);
  __syncthreads();

  if (t < 64) {
    float s = 0.f;
#pragma unroll
    for (int r = 0; r < 16; ++r) s += Yt[r * 64 + t];
    atomAddF(&colsum[t], s);
  }
}

__global__ void semantic_kernel(const float* __restrict__ cs0, const float* __restrict__ cs1,
                                const float* __restrict__ q, float* __restrict__ attn) {
  float s0 = 0.f, s1 = 0.f;
  for (int c = 0; c < 64; ++c) { s0 += q[c] * cs0[c]; s1 += q[c] * cs1[c]; }
  s0 *= (1.0f / NN); s1 *= (1.0f / NN);
  const float m = fmaxf(s0, s1);
  const float e0 = __expf(s0 - m), e1 = __expf(s1 - m);
  attn[0] = e0 / (e0 + e1);
  attn[1] = e1 / (e0 + e1);
}

__global__ void fuse_relu_kernel(const float* __restrict__ st0, const float* __restrict__ st1,
                                 const float* __restrict__ attn, float* __restrict__ H, int n) {
  int i = blockIdx.x * blockDim.x + threadIdx.x;
  if (i >= n) return;
  const float v = attn[0] * st0[i] + attn[1] * st1[i];
  H[i] = v > 0.f ? v : 0.f;
}

// ---------------------------------------------------------------------------
// Final linear: OUT = H @ W^T + b   (N x 64) x (16 x 64) -> (N x 16)
// Block = 128 threads (4 waves), each wave = one 16x16 tile; block covers 64 rows.
// ---------------------------------------------------------------------------
__global__ void __launch_bounds__(128)
lin_out_kernel(const float* __restrict__ H, const float* __restrict__ W,
               const float* __restrict__ bias, float* __restrict__ OUT, int N) {
  __shared__ __bf16 As[64 * 64];
  __shared__ __bf16 Ws[16 * 64];

  const int t = threadIdx.x, lane = t & 31, wave = t >> 5;
  const int rowblk = blockIdx.x * 64;

  for (int i = t; i < 16 * 64; i += 128) Ws[i] = (__bf16)W[i];
  for (int i = t; i < 64 * 64; i += 128) {
    const int row = rowblk + (i >> 6);
    As[i] = (__bf16)(row < N ? H[row * 64 + (i & 63)] : 0.f);
  }
  __syncthreads();

  const int row0 = rowblk + wave * 16;
  if (row0 >= N) return;

  floatx8 acc = {};
#pragma unroll
  for (int kb = 0; kb < 64; kb += 32) {
    bf16x16 a = frag_a(As + wave * 16 * 64, 64, kb, lane);
    bf16x16 b = frag_b(Ws, 64, kb, 0, lane);
    acc = __builtin_amdgcn_wmma_f32_16x16x32_bf16(false, a, false, b, (short)0, acc,
                                                  false, false);
  }
  const int n = lane & 15, mo = (lane >> 4) << 3;
#pragma unroll
  for (int r = 0; r < 8; ++r)
    OUT[(row0 + mo + r) * 16 + n] = acc[r] + bias[n];
}

// ---------------------------------------------------------------------------
// Host launcher
// ---------------------------------------------------------------------------
extern "C" void kernel_launch(void* const* d_in, const int* in_sizes, int n_in,
                              void* d_out, int out_size, void* d_ws, size_t ws_size,
                              hipStream_t stream) {
  (void)in_sizes; (void)n_in; (void)out_size; (void)ws_size;

  // setup_inputs() dict flattened in insertion order:
  const float* x    = (const float*)d_in[0];
  const int*   e_sp = (const int*)d_in[1];   // [2, E]
  const int*   e_si = (const int*)d_in[2];   // [2, E]
  // conv1: W, b, att_src0, att_src1, att_dst0, att_dst1, kW, kb, q
  const float* c1W  = (const float*)d_in[3];
  const float* c1b  = (const float*)d_in[4];
  const float* c1s0 = (const float*)d_in[5];
  const float* c1s1 = (const float*)d_in[6];
  const float* c1d0 = (const float*)d_in[7];
  const float* c1d1 = (const float*)d_in[8];
  const float* c1kW = (const float*)d_in[9];
  const float* c1kb = (const float*)d_in[10];
  const float* c1q  = (const float*)d_in[11];
  const float* c2W  = (const float*)d_in[12];
  const float* c2b  = (const float*)d_in[13];
  const float* c2s0 = (const float*)d_in[14];
  const float* c2s1 = (const float*)d_in[15];
  const float* c2d0 = (const float*)d_in[16];
  const float* c2d1 = (const float*)d_in[17];
  const float* c2kW = (const float*)d_in[18];
  const float* c2kb = (const float*)d_in[19];
  const float* c2q  = (const float*)d_in[20];
  const float* lW   = (const float*)d_in[21];
  const float* lb   = (const float*)d_in[22];

  float* ws = (float*)d_ws;
  float* xp     = ws;                 ws += NN * 64;   // projected nodes
  float* hbuf   = ws;                 ws += NN * 64;   // layer output
  float* st0    = ws;                 ws += NN * 64;   // metapath 0 (agg -> st)
  float* st1    = ws;                 ws += NN * 64;   // metapath 1
  float* asrc0  = ws;                 ws += NN * 8;
  float* asrc1  = ws;                 ws += NN * 8;
  float* adst0  = ws;                 ws += NN * 8;
  float* adst1  = ws;                 ws += NN * 8;
  float* segmax = ws;                 ws += NN * 8;
  float* den    = ws;                 ws += NN * 8;
  float* colsum = ws;                 ws += 128;
  float* attn   = ws;                 ws += 2;

  const int nTiles   = NN / 16;                 // 3125
  const int edgeGrid = (NE * 8 + 255) / 256;    // 25000
  const int nodGrid  = (NN * 64 + 255) / 256;   // 12500
  const int fillGrid = (NN * 8 + 255) / 256;

  auto run_layer = [&](const float* Xin, int Kdim, const float* W, const float* b,
                       const float* s0, const float* s1, const float* dd0, const float* dd1,
                       const float* kW, const float* kb, const float* q, float* Hout) {
    if (Kdim == 128)
      proj_att_kernel<128><<<nTiles, 128, 0, stream>>>(Xin, W, b, s0, s1, dd0, dd1, xp,
                                                       asrc0, asrc1, adst0, adst1);
    else
      proj_att_kernel<64><<<nTiles, 128, 0, stream>>>(Xin, W, b, s0, s1, dd0, dd1, xp,
                                                      asrc0, asrc1, adst0, adst1);

    const int* eptr[2] = {e_sp, e_si};
    float* stp[2]      = {st0, st1};
    const float* asp[2] = {asrc0, asrc1};
    const float* adp[2] = {adst0, adst1};

    for (int m = 0; m < 2; ++m) {
      const int* src = eptr[m];
      const int* dst = eptr[m] + NE;
      fill_kernel<<<fillGrid, 256, 0, stream>>>(segmax, -3.0e38f, NN * 8);
      hipMemsetAsync(den, 0, NN * 8 * sizeof(float), stream);
      hipMemsetAsync(stp[m], 0, NN * 64 * sizeof(float), stream);
      edge_max_kernel<<<edgeGrid, 256, 0, stream>>>(src, dst, asp[m], adp[m], segmax, NE);
      edge_agg_kernel<<<edgeGrid, 256, 0, stream>>>(src, dst, asp[m], adp[m], segmax, xp,
                                                    den, stp[m], NE);
      norm_relu_kernel<<<nodGrid, 256, 0, stream>>>(den, stp[m], NN * 64);
    }

    hipMemsetAsync(colsum, 0, 128 * sizeof(float), stream);
    tanh_colsum_kernel<<<nTiles, 128, 0, stream>>>(st0, kW, kb, colsum);
    tanh_colsum_kernel<<<nTiles, 128, 0, stream>>>(st1, kW, kb, colsum + 64);
    semantic_kernel<<<1, 1, 0, stream>>>(colsum, colsum + 64, q, attn);
    fuse_relu_kernel<<<nodGrid, 256, 0, stream>>>(st0, st1, attn, Hout, NN * 64);
  };

  // Layer 1: x [N,128] -> hbuf [N,64]
  run_layer(x, 128, c1W, c1b, c1s0, c1s1, c1d0, c1d1, c1kW, c1kb, c1q, hbuf);
  // Layer 2: hbuf [N,64] -> xp reused as h2 [N,64]
  float* h2 = st0;  // st0 is free after layer-2 fusion writes... but fusion reads st0.
  h2 = xp;          // xp is free once layer-2 edge aggregation is done; fuse writes here.
  run_layer(hbuf, 64, c2W, c2b, c2s0, c2s1, c2d0, c2d1, c2kW, c2kb, c2q, h2);

  // Final linear: h2 [N,64] -> out [N,16]
  lin_out_kernel<<<(NN + 63) / 64, 128, 0, stream>>>(h2, lW, lb, (float*)d_out, NN);
}
